// _SyntheticSparseMoeBlock_45140106281246
// MI455X (gfx1250) — compile-verified, matrix operand order checked by reading, not run on hardware
//
#include <hip/hip_runtime.h>
#include <hip/hip_bf16.h>

// ---------------------------------------------------------------------------
// MoE block for MI455X (gfx1250, wave32): router + 8 experts, bf16 WMMA path.
//   out[t] = sum_e w[t,e] * ( silu(x gu_g^T) * (x gu_u^T) ) dp^T
// HBM-bound on streaming ~805MB of fp32 expert weights (~37us at 23.3TB/s).
// A-operands (already bf16) are staged to LDS with GLOBAL_LOAD_ASYNC_TO_LDS
// (ASYNCcnt); B weights are read once as fp32, packed to bf16 quads, and
// ds_store'd in WMMA fragment order; v_wmma_f32_16x16x32_bf16 with f32 accum.
// ---------------------------------------------------------------------------

typedef __bf16 v16bf __attribute__((ext_vector_type(16)));
typedef float  v8f   __attribute__((ext_vector_type(8)));
typedef unsigned long long u64;

constexpr int Tt = 8192;   // tokens = B*S
constexpr int Hd = 2048;   // hidden dim
constexpr int Id = 4096;   // intermediate dim
constexpr int NE = 8;      // experts

// Element index inside a 16x32 bf16 WMMA fragment tile (512 elements),
// matching the ISA 16-bit A/B VGPR layout:
//   lane = m + 16*((k>>3)&1), element = ((k>>4)<<3) | (k&7)
// For k % 4 == 0 the 4 elements k..k+3 are contiguous (8B aligned).
__device__ __forceinline__ int frag_idx(int m, int k) {
    int lane = m + (((k >> 3) & 1) << 4);
    int e    = ((k >> 4) << 3) | (k & 7);
    return lane * 16 + e;
}

__device__ __forceinline__ u64 pack4_bf16(float4 v) {
    union { __bf16 h[4]; u64 u; } p;
    p.h[0] = (__bf16)v.x; p.h[1] = (__bf16)v.y;
    p.h[2] = (__bf16)v.z; p.h[3] = (__bf16)v.w;
    return p.u;
}

// silu(g) = g * rcp(1 + exp(-g)); v_rcp_f32 instead of the IEEE divide macro.
__device__ __forceinline__ float silu_fast(float g) {
    return g * __builtin_amdgcn_rcpf(1.0f + __expf(-g));
}

// Async global->LDS copy of 8 bytes per lane (GV mode, ASYNCcnt-tracked).
__device__ __forceinline__ void async_copy_b64(unsigned lds_off, u64 gaddr) {
    asm volatile("global_load_async_to_lds_b64 %0, %1, off"
                 :: "v"(lds_off), "v"(gaddr) : "memory");
}
__device__ __forceinline__ void wait_asynccnt0() {
    asm volatile("s_wait_asynccnt 0x0" ::: "memory");
}

// ---------------------------------------------------------------------------
// Cast x [T,H] fp32 -> bf16 once (coalesced b128 loads, b64 stores).
// ---------------------------------------------------------------------------
__global__ void cast_x_kernel(const float* __restrict__ x, __bf16* __restrict__ xb) {
    size_t i = ((size_t)blockIdx.x * 256 + threadIdx.x) * 4;
    float4 v = *reinterpret_cast<const float4*>(&x[i]);
    *reinterpret_cast<u64*>(&xb[i]) = pack4_bf16(v);
}

// ---------------------------------------------------------------------------
// Router: logits = x @ gate_weight^T, softmax over 8, top-2, renormalize.
// One wave (32 lanes) per token; gate weights cached in LDS (64KB).
// ---------------------------------------------------------------------------
__global__ void router_kernel(const float* __restrict__ x,
                              const float* __restrict__ gw,
                              float* __restrict__ combine) {
    __shared__ float Gs[NE * Hd];
    const int tid = threadIdx.x;
    for (int idx = tid; idx < NE * Hd; idx += 256) Gs[idx] = gw[idx];
    __syncthreads();

    const int wave = tid >> 5;
    const int lane = tid & 31;
    const int t = blockIdx.x * 8 + wave;

    float acc[NE];
#pragma unroll
    for (int e = 0; e < NE; ++e) acc[e] = 0.0f;

    const float* xr = x + (size_t)t * Hd;
    for (int h = lane; h < Hd; h += 32) {
        float xv = xr[h];
#pragma unroll
        for (int e = 0; e < NE; ++e) acc[e] = fmaf(xv, Gs[e * Hd + h], acc[e]);
    }
#pragma unroll
    for (int e = 0; e < NE; ++e) {
#pragma unroll
        for (int off = 16; off > 0; off >>= 1)
            acc[e] += __shfl_xor(acc[e], off, 32);
    }

    if (lane < NE) {
        float mx = acc[0];
#pragma unroll
        for (int e = 1; e < NE; ++e) mx = fmaxf(mx, acc[e]);
        float p[NE];
#pragma unroll
        for (int e = 0; e < NE; ++e) p[e] = __expf(acc[e] - mx);
        int i1 = 0;
#pragma unroll
        for (int e = 1; e < NE; ++e) if (p[e] > p[i1]) i1 = e;
        int i2 = (i1 == 0) ? 1 : 0;
#pragma unroll
        for (int e = 0; e < NE; ++e) if (e != i1 && p[e] > p[i2]) i2 = e;
        float inv = 1.0f / (p[i1] + p[i2]);
        float wv = (lane == i1) ? p[i1] * inv : ((lane == i2) ? p[i2] * inv : 0.0f);
        combine[(size_t)t * NE + lane] = wv;
    }
}

// ---------------------------------------------------------------------------
// Expert stage 1: hid = silu(xb @ gu_gate^T) * (xb @ gu_up^T)   [T, I] bf16
// Block tile: 128 tokens x 64 i-cols; 8 waves, each a 16-row strip with
// 4 gate + 4 up f32 accumulator tiles. K-loop over H in 32-wide slabs.
// ---------------------------------------------------------------------------
__global__ void gate_up_kernel(const __bf16* __restrict__ xb,   // [T, H] bf16
                               const float* __restrict__ gup,   // [2I, H] one expert
                               __bf16* __restrict__ hid) {      // [T, I]
    __shared__ __align__(32) __bf16 As[8][512];
    __shared__ __align__(32) __bf16 Bg[4][512];
    __shared__ __align__(32) __bf16 Bu[4][512];

    const int tid = threadIdx.x;
    const int wave = tid >> 5;
    const int lane = tid & 31;
    const int t0 = blockIdx.x * 128;
    const int i0 = blockIdx.y * 64;

    // --- A staging via async-to-LDS: per-thread 4 quads (8B each) ---
    u64 asrc[4]; unsigned adst[4];
#pragma unroll
    for (int j = 0; j < 4; ++j) {
        int idx = j * 256 + tid;
        int m = idx >> 3, k = (idx & 7) * 4;
        asrc[j] = (u64)(uintptr_t)(xb + (size_t)(t0 + m) * Hd + k);
        adst[j] = (unsigned)(uintptr_t)&As[m >> 4][frag_idx(m & 15, k)];
    }
    // --- B staging via VGPRs (fp32 -> bf16 pack): 2+2 quads per thread ---
    const float4* bgsrc[2]; const float4* busrc[2]; u64* bgdst[2]; u64* budst[2];
#pragma unroll
    for (int j = 0; j < 2; ++j) {
        int idx = j * 256 + tid;
        int n = idx >> 3, k = (idx & 7) * 4;
        bgsrc[j] = reinterpret_cast<const float4*>(gup + (size_t)(i0 + n) * Hd + k);
        busrc[j] = reinterpret_cast<const float4*>(gup + (size_t)(Id + i0 + n) * Hd + k);
        bgdst[j] = reinterpret_cast<u64*>(&Bg[n >> 4][frag_idx(n & 15, k)]);
        budst[j] = reinterpret_cast<u64*>(&Bu[n >> 4][frag_idx(n & 15, k)]);
    }

    v8f gacc[4] = {};
    v8f uacc[4] = {};

    for (int k0 = 0; k0 < Hd; k0 += 32) {
        __syncthreads();
#pragma unroll
        for (int j = 0; j < 4; ++j) {            // A: async bf16 quad copy
            async_copy_b64(adst[j], asrc[j]);
            asrc[j] += 64;                       // +32 bf16
        }
#pragma unroll
        for (int j = 0; j < 2; ++j) {            // B: fp32 quad -> bf16 quad
            *bgdst[j] = pack4_bf16(*bgsrc[j]);
            *budst[j] = pack4_bf16(*busrc[j]);
            bgsrc[j] += 8;                       // +32 floats
            busrc[j] += 8;
        }
        __builtin_prefetch(bgsrc[0] + 8, 0, 0);  // next k-slab of weight stream
        __builtin_prefetch(busrc[0] + 8, 0, 0);
        wait_asynccnt0();
        __syncthreads();

        v16bf a = *reinterpret_cast<const v16bf*>(&As[wave][lane * 16]);

        v16bf bf[4];
#pragma unroll
        for (int c = 0; c < 4; ++c)
            bf[c] = *reinterpret_cast<const v16bf*>(&Bg[c][lane * 16]);
#pragma unroll
        for (int c = 0; c < 4; ++c)
            gacc[c] = __builtin_amdgcn_wmma_f32_16x16x32_bf16(
                false, a, false, bf[c], (short)0, gacc[c], false, false);

#pragma unroll
        for (int c = 0; c < 4; ++c)
            bf[c] = *reinterpret_cast<const v16bf*>(&Bu[c][lane * 16]);
#pragma unroll
        for (int c = 0; c < 4; ++c)
            uacc[c] = __builtin_amdgcn_wmma_f32_16x16x32_bf16(
                false, a, false, bf[c], (short)0, uacc[c], false, false);
    }

    // epilogue: silu(g) * u -> bf16 hidden activations
    const int hl = lane >> 4;
    const int nl = lane & 15;
#pragma unroll
    for (int c = 0; c < 4; ++c) {
#pragma unroll
        for (int r = 0; r < 8; ++r) {
            float act = silu_fast(gacc[c][r]) * uacc[c][r];
            int t = t0 + wave * 16 + r + 8 * hl;
            int i = i0 + c * 16 + nl;
            hid[(size_t)t * Id + i] = (__bf16)act;
        }
    }
}

// ---------------------------------------------------------------------------
// Expert stage 2: out (+)= w[t,e] * (hid @ dp^T)    [T, H]
// Block tile: 128 tokens x 64 h-cols; K-loop over I. Sequential expert
// launches own disjoint out tiles -> plain read-modify-write, no atomics.
// ---------------------------------------------------------------------------
__global__ void down_kernel(const __bf16* __restrict__ hid,   // [T, I] bf16
                            const float* __restrict__ dp,     // [H, I] one expert
                            const float* __restrict__ combine,// [T, E]
                            float* __restrict__ out,          // [T, H]
                            int eidx, int first) {
    __shared__ __align__(32) __bf16 As[8][512];
    __shared__ __align__(32) __bf16 Bs[4][512];

    const int tid = threadIdx.x;
    const int wave = tid >> 5;
    const int lane = tid & 31;
    const int t0 = blockIdx.x * 128;
    const int h0 = blockIdx.y * 64;

    u64 asrc[4]; unsigned adst[4];
#pragma unroll
    for (int j = 0; j < 4; ++j) {
        int idx = j * 256 + tid;
        int m = idx >> 3, k = (idx & 7) * 4;
        asrc[j] = (u64)(uintptr_t)(hid + (size_t)(t0 + m) * Id + k);
        adst[j] = (unsigned)(uintptr_t)&As[m >> 4][frag_idx(m & 15, k)];
    }
    const float4* bsrc[2]; u64* bdst[2];
#pragma unroll
    for (int j = 0; j < 2; ++j) {
        int idx = j * 256 + tid;
        int n = idx >> 3, k = (idx & 7) * 4;
        bsrc[j] = reinterpret_cast<const float4*>(dp + (size_t)(h0 + n) * Id + k);
        bdst[j] = reinterpret_cast<u64*>(&Bs[n >> 4][frag_idx(n & 15, k)]);
    }

    v8f acc[4] = {};

    for (int k0 = 0; k0 < Id; k0 += 32) {
        __syncthreads();
#pragma unroll
        for (int j = 0; j < 4; ++j) {
            async_copy_b64(adst[j], asrc[j]);
            asrc[j] += 64;
        }
#pragma unroll
        for (int j = 0; j < 2; ++j) {
            *bdst[j] = pack4_bf16(*bsrc[j]);
            bsrc[j] += 8;
        }
        __builtin_prefetch(bsrc[0] + 8, 0, 0);
        wait_asynccnt0();
        __syncthreads();

        v16bf a = *reinterpret_cast<const v16bf*>(&As[wave][lane * 16]);

        v16bf bf[4];
#pragma unroll
        for (int c = 0; c < 4; ++c)
            bf[c] = *reinterpret_cast<const v16bf*>(&Bs[c][lane * 16]);
#pragma unroll
        for (int c = 0; c < 4; ++c)
            acc[c] = __builtin_amdgcn_wmma_f32_16x16x32_bf16(
                false, a, false, bf[c], (short)0, acc[c], false, false);
    }

    const int hl = lane >> 4;
    const int nl = lane & 15;
#pragma unroll
    for (int r = 0; r < 8; ++r) {
        int t = t0 + wave * 16 + r + 8 * hl;
        float w = combine[(size_t)t * NE + eidx];
#pragma unroll
        for (int c = 0; c < 4; ++c) {
            int h = h0 + c * 16 + nl;
            size_t o = (size_t)t * Hd + h;
            float y = w * acc[c][r];
            out[o] = first ? y : (out[o] + y);
        }
    }
}

// ---------------------------------------------------------------------------
extern "C" void kernel_launch(void* const* d_in, const int* in_sizes, int n_in,
                              void* d_out, int out_size, void* d_ws, size_t ws_size,
                              hipStream_t stream) {
    const float* x   = (const float*)d_in[0];  // [T, H]
    const float* gw  = (const float*)d_in[1];  // [E, H]
    const float* gup = (const float*)d_in[2];  // [E, 2I, H]
    const float* dpw = (const float*)d_in[3];  // [E, H, I]
    float* out = (float*)d_out;                // [T, H]

    // workspace layout: combine_w [T,E] f32 | xb [T,H] bf16 | hid [T,I] bf16
    char* ws = (char*)d_ws;
    float*  combine = (float*)ws;
    __bf16* xb  = (__bf16*)(ws + (size_t)Tt * NE * sizeof(float));
    __bf16* hid = (__bf16*)(ws + (size_t)Tt * NE * sizeof(float)
                               + (size_t)Tt * Hd * sizeof(__bf16));

    cast_x_kernel<<<(Tt * (size_t)Hd) / (256 * 4), 256, 0, stream>>>(x, xb);
    router_kernel<<<Tt / 8, 256, 0, stream>>>(x, gw, combine);

    for (int e = 0; e < NE; ++e) {
        gate_up_kernel<<<dim3(Tt / 128, Id / 64), 256, 0, stream>>>(
            xb, gup + (size_t)e * 2 * Id * Hd, hid);
        down_kernel<<<dim3(Tt / 128, Hd / 64), 256, 0, stream>>>(
            hid, dpw + (size_t)e * (size_t)Hd * Id, combine, out, e, e == 0 ? 1 : 0);
    }
}
